// LinearAttention_67138928771246
// MI455X (gfx1250) — matrix operand order repaired and checked by reading, not compiled, not run on hardware
//
#include <hip/hip_runtime.h>
#include <hip/hip_bf16.h>
#include <math.h>

typedef __attribute__((ext_vector_type(16))) _Float16 v16h;
typedef __attribute__((ext_vector_type(8)))  _Float16 v8h;
typedef __attribute__((ext_vector_type(8)))  float    v8f;
typedef int v4i __attribute__((vector_size(16)));   // b128 payload type for async builtins

#define AS3 __attribute__((address_space(3)))
#define AS1 __attribute__((address_space(1)))

#define D_MODELC 1024
#define N_HEADC  16
#define D_HEADC  64
#define D_KMAPC  128
#define BATCHC   4
#define SEQC     4096
#define BLC      (BATCHC*SEQC)   /* 16384 */
#define TWO_M    (2*D_KMAPC)     /* 256   */

#define BM 128
#define BN 64
#define BK 32
#define NTHREADS 256

// ---------------------------------------------------------------------------
// CDNA5 async global->LDS copy (ASYNCcnt-tracked), with inline-asm fallback.
// ---------------------------------------------------------------------------
__device__ __forceinline__ void async_b128(const _Float16* g, _Float16* l) {
#if __has_builtin(__builtin_amdgcn_global_load_async_to_lds_b128)
    __builtin_amdgcn_global_load_async_to_lds_b128(
        (AS1 v4i*)g, (AS3 v4i*)l, 0, 0);
#else
    unsigned lds_off = (unsigned)(unsigned long long)((AS3 void*)l);
    asm volatile("global_load_async_to_lds_b128 %0, %1, off"
                 :: "v"(lds_off), "v"(g) : "memory");
#endif
}

template <int N>
__device__ __forceinline__ void wait_asynccnt() {
#if __has_builtin(__builtin_amdgcn_s_wait_asynccnt)
    __builtin_amdgcn_s_wait_asynccnt(N);
#else
    asm volatile("s_wait_asynccnt %0" :: "i"(N));
#endif
}

// ---------------------------------------------------------------------------
// Fragment loaders matching the CDNA5 documented wave32 VGPR layouts.
// A (16-bit, 16x32 MxK): lane<16 -> row=lane,    K = {0..7, 16..23};
//                        lane>=16 -> row=lane-16, K = {8..15, 24..31}.
// LDS As[m][k] row-contiguous => two contiguous b128 reads.
// ---------------------------------------------------------------------------
__device__ __forceinline__ v16h load_frag_a(const _Float16* As, int m_base, int lane) {
    const int row = m_base + (lane & 15);
    const int h   = lane >> 4;
    const _Float16* p = As + row * BK;
    v8h lo = *(const v8h*)(p + h * 8);
    v8h hi = *(const v8h*)(p + 16 + h * 8);
    v16h a;
#pragma unroll
    for (int i = 0; i < 8; ++i) { a[i] = lo[i]; a[i + 8] = hi[i]; }
    return a;
}

// B (16-bit, 32x16 KxN): lane<16 -> N=lane, K={0..7,16..23}; lane>=16 -> N=lane-16, K={8..15,24..31}.
// LDS Bs[n][k] => two contiguous b128 reads.
__device__ __forceinline__ v16h load_frag_b(const _Float16* Bs, int n_base, int lane) {
    const int n  = n_base + (lane & 15);
    const int kb = (lane < 16) ? 0 : 8;
    const _Float16* p = Bs + n * BK + kb;
    v8h lo = *(const v8h*)(p);
    v8h hi = *(const v8h*)(p + 16);
    v16h b;
#pragma unroll
    for (int i = 0; i < 8; ++i) { b[i] = lo[i]; b[i + 8] = hi[i]; }
    return b;
}

// ---------------------------------------------------------------------------
// Generic batched wave-tiled WMMA GEMM:  C[M,N] = A[M,K] * Bt^T  (+ epilogue)
//   A: f16, rows contiguous over K, row stride lda.
//   Bt: f16 "transposed-B" layout Bt[n][k], row stride ldb  -> tiles are
//       contiguous 16B chunks => pure async global->LDS staging, double-buffered.
//   Epilogue MODE:
//     0: f16 C[row,col] = v + bias        (Q/K projections)
//     1: f32 C[row,col] = v + bias        (final projection -> d_out)
//     2: f16 C[row,col] = v               (out-einsum)
//     3: Fourier std:  qp[row,n]=cos(v)/sqrt(128), qp[row,128+n]=sin(v)/sqrt(128)
//     4: f16 V^T:      vT[b,h,d,l] = v + bias   (row=(b,l), col=(h,d))
//     5: Fourier k'^T: kpT[b,h,m,l] (and m+128)  (row=(b,l,h), col=m)
//     6: f16 C^T:      C[cz + col*ldc + row] = v (kv -> kvT[b,h,d,m])
// grid.z batching: z -> (bz=z/HDIV, hz=z%HDIV) with per-operand strides.
// ---------------------------------------------------------------------------
template <int MODE>
__global__ __launch_bounds__(NTHREADS)
void gemm_wmma(const _Float16* __restrict__ A, long lda,
               const _Float16* __restrict__ Bt, long ldb,
               const float* __restrict__ bias,
               void* __restrict__ Cout, long ldc, int K,
               long sAB, long sAH, long sBB, long sBH, long sCB, long sCH,
               int HDIV) {
    __shared__ _Float16 As[2][BM * BK];
    __shared__ _Float16 Bs[2][BN * BK];

    const int z  = blockIdx.z;
    const int bz = (HDIV > 1) ? (z / HDIV) : z;
    const int hz = (HDIV > 1) ? (z % HDIV) : 0;

    const _Float16* Ab = A + (long)bz * sAB + (long)hz * sAH + (long)blockIdx.x * BM * lda;
    const _Float16* Bb = Bt + (long)bz * sBB + (long)hz * sBH + (long)blockIdx.y * BN * ldb;
    const long crow0 = (long)blockIdx.x * BM;
    const long ccol0 = (long)blockIdx.y * BN;
    const long cz    = (long)bz * sCB + (long)hz * sCH;

    const int tid  = threadIdx.x;
    const int lane = tid & 31;
    const int wid  = tid >> 5;

    // Per tile, each thread issues exactly 3 async b128 copies (2 for A, 1 for B).
    auto load_tile = [&](int buf, int k0) {
#pragma unroll
        for (int i = tid; i < (BM * BK / 8); i += NTHREADS) {   // 512 chunks
            const int row = i >> 2;
            const int c8  = (i & 3) * 8;
            async_b128(Ab + (long)row * lda + k0 + c8, &As[buf][row * BK + c8]);
        }
        {
            const int n  = tid >> 2;                            // 256 chunks
            const int c8 = (tid & 3) * 8;
            async_b128(Bb + (long)n * ldb + k0 + c8, &Bs[buf][n * BK + c8]);
        }
    };

    v8f acc[4];
#pragma unroll
    for (int t = 0; t < 4; ++t)
#pragma unroll
        for (int r = 0; r < 8; ++r) acc[t][r] = 0.0f;

    const int nsteps = K / BK;
    load_tile(0, 0);
    for (int s = 0; s < nsteps; ++s) {
        const int cur = s & 1;
        if (s + 1 < nsteps) {
            load_tile(cur ^ 1, (s + 1) * BK);   // prefetch next tile
            wait_asynccnt<3>();                 // retire previous tile only
        } else {
            wait_asynccnt<0>();
        }
        __syncthreads();

        const v16h a = load_frag_a(&As[cur][0], wid * 16, lane);
#pragma unroll
        for (int t = 0; t < 4; ++t) {
            const v16h b = load_frag_b(&Bs[cur][0], t * 16, lane);
            acc[t] = __builtin_amdgcn_wmma_f32_16x16x32_f16(
                false, a, false, b, (short)0, acc[t], false, false);
        }
        __syncthreads();   // protect buffer 'cur' before it is refilled
    }

    // Epilogue.  C frag layout: VGPR r -> row (lane<16 ? r : 8+r), col = lane&15.
    const int  rlo    = (lane < 16) ? 0 : 8;
    const int  col_in = lane & 15;
    const long mbase  = crow0 + wid * 16;
#pragma unroll
    for (int t = 0; t < 4; ++t) {
        const long ncol = ccol0 + t * 16 + col_in;
#pragma unroll
        for (int r = 0; r < 8; ++r) {
            const long row = mbase + rlo + r;
            const float v = acc[t][r];
            if (MODE == 0) {
                ((_Float16*)Cout)[cz + row * ldc + ncol] = (_Float16)(v + bias[ncol]);
            } else if (MODE == 1) {
                ((float*)Cout)[cz + row * ldc + ncol] = v + bias[ncol];
            } else if (MODE == 2) {
                ((_Float16*)Cout)[cz + row * ldc + ncol] = (_Float16)v;
            } else if (MODE == 3) {
                const float s = 0.08838834764831845f;  // 1/sqrt(128)
                _Float16* o = (_Float16*)Cout;
                o[row * TWO_M + ncol]            = (_Float16)(cosf(v) * s);
                o[row * TWO_M + D_KMAPC + ncol]  = (_Float16)(sinf(v) * s);
            } else if (MODE == 4) {
                const long b = row >> 12, l = row & (SEQC - 1);
                const long h = ncol >> 6, d = ncol & 63;
                ((_Float16*)Cout)[((b * N_HEADC + h) * D_HEADC + d) * SEQC + l] =
                    (_Float16)(v + bias[ncol]);
            } else if (MODE == 5) {
                const float s = 0.08838834764831845f;
                const long h  = row & 15, bl = row >> 4;
                const long b  = bl >> 12, l = bl & (SEQC - 1);
                _Float16* o = (_Float16*)Cout;
                const long base = (b * N_HEADC + h) * TWO_M;
                o[(base + ncol) * SEQC + l]           = (_Float16)(cosf(v) * s);
                o[(base + D_KMAPC + ncol) * SEQC + l] = (_Float16)(sinf(v) * s);
            } else {  // MODE 6: transposed store
                ((_Float16*)Cout)[cz + ncol * ldc + row] = (_Float16)v;
            }
        }
    }
}

// ---------------------------------------------------------------------------
// Elementwise f32 -> f16 cast (optionally scaled).
// ---------------------------------------------------------------------------
__global__ void cast_f32_f16(const float* __restrict__ in, _Float16* __restrict__ out,
                             float scale, long n) {
    const long i = (long)blockIdx.x * blockDim.x + threadIdx.x;
    if (i < n) out[i] = (_Float16)(in[i] * scale);
}

// projT[m][d] = proj[d][m] * 1/sqrt(D_HEAD)   (128 x 64 from 64 x 128)
__global__ void transpose_proj(const float* __restrict__ proj, _Float16* __restrict__ projT) {
    const int i = blockIdx.x * blockDim.x + threadIdx.x;   // 8192 elems
    if (i < D_KMAPC * D_HEADC) {
        const int m = i >> 6, d = i & 63;
        projT[m * D_HEADC + d] = (_Float16)(proj[d * D_KMAPC + m] * 0.125f);
    }
}

// ---------------------------------------------------------------------------
extern "C" void kernel_launch(void* const* d_in, const int* in_sizes, int n_in,
                              void* d_out, int out_size, void* d_ws, size_t ws_size,
                              hipStream_t stream) {
    (void)in_sizes; (void)n_in; (void)out_size; (void)ws_size;
    const float* x    = (const float*)d_in[0];
    const float* proj = (const float*)d_in[1];
    const float* Wq   = (const float*)d_in[2];
    const float* bq   = (const float*)d_in[3];
    const float* Wk   = (const float*)d_in[4];
    const float* bk   = (const float*)d_in[5];
    const float* Wv   = (const float*)d_in[6];
    const float* bv   = (const float*)d_in[7];
    const float* Wo   = (const float*)d_in[8];
    const float* bo   = (const float*)d_in[9];

    char* ws = (char*)d_ws;
    size_t off = 0;
    auto alloc = [&](size_t bytes) -> char* {
        char* p = ws + off;
        off += (bytes + 255) & ~(size_t)255;
        return p;
    };

    const size_t act = (size_t)BLC * D_MODELC * 2;                  // 32 MB f16
    _Float16* xh    = (_Float16*)alloc(act);
    _Float16* Wqh   = (_Float16*)alloc((size_t)D_MODELC * D_MODELC * 2);
    _Float16* Wkh   = (_Float16*)alloc((size_t)D_MODELC * D_MODELC * 2);
    _Float16* Wvh   = (_Float16*)alloc((size_t)D_MODELC * D_MODELC * 2);
    _Float16* Woh   = (_Float16*)alloc((size_t)D_MODELC * D_MODELC * 2);
    _Float16* projT = (_Float16*)alloc((size_t)D_KMAPC * D_HEADC * 2);
    _Float16* qh    = (_Float16*)alloc(act);
    _Float16* kh    = (_Float16*)alloc(act);
    _Float16* vT    = (_Float16*)alloc(act);                        // [B,H,64,L]
    _Float16* qp    = (_Float16*)alloc((size_t)BLC * N_HEADC * TWO_M * 2);  // [B,L,H,256]
    _Float16* kpT   = (_Float16*)alloc((size_t)BLC * N_HEADC * TWO_M * 2);  // [B,H,256,L]
    _Float16* kvT   = (_Float16*)alloc((size_t)BATCHC * N_HEADC * D_HEADC * TWO_M * 2); // [B,H,64,256]
    _Float16* attn  = (_Float16*)alloc(act);                        // [B,L,1024]

    // 1) f16 casts: x, and the four weights (W[n][k] row-major IS the Bt layout).
    {
        const long n = (long)BLC * D_MODELC;
        cast_f32_f16<<<dim3((unsigned)((n + 255) / 256)), 256, 0, stream>>>(x, xh, 1.0f, n);
        const long nw = (long)D_MODELC * D_MODELC;
        dim3 gw((unsigned)((nw + 255) / 256));
        cast_f32_f16<<<gw, 256, 0, stream>>>(Wq, Wqh, 1.0f, nw);
        cast_f32_f16<<<gw, 256, 0, stream>>>(Wk, Wkh, 1.0f, nw);
        cast_f32_f16<<<gw, 256, 0, stream>>>(Wv, Wvh, 1.0f, nw);
        cast_f32_f16<<<gw, 256, 0, stream>>>(Wo, Woh, 1.0f, nw);
        transpose_proj<<<32, 256, 0, stream>>>(proj, projT);
    }
    // 2) Q/K projections (f16 C), V projection written transposed per head.
    {
        dim3 g(BLC / BM, D_MODELC / BN, 1);
        gemm_wmma<0><<<g, NTHREADS, 0, stream>>>(xh, D_MODELC, Wqh, D_MODELC, bq, qh,
                                                 D_MODELC, D_MODELC, 0, 0, 0, 0, 0, 0, 1);
        gemm_wmma<0><<<g, NTHREADS, 0, stream>>>(xh, D_MODELC, Wkh, D_MODELC, bk, kh,
                                                 D_MODELC, D_MODELC, 0, 0, 0, 0, 0, 0, 1);
        gemm_wmma<4><<<g, NTHREADS, 0, stream>>>(xh, D_MODELC, Wvh, D_MODELC, bv, vT,
                                                 0, D_MODELC, 0, 0, 0, 0, 0, 0, 1);
    }
    // 3) Fourier features (rows = BL*H = 262144, K=64, N=128):
    //    q' standard layout; k' written transposed [B,H,256,L].
    {
        dim3 g((BLC * N_HEADC) / BM, D_KMAPC / BN, 1);
        gemm_wmma<3><<<g, NTHREADS, 0, stream>>>(qh, D_HEADC, projT, D_HEADC, nullptr, qp,
                                                 TWO_M, D_HEADC, 0, 0, 0, 0, 0, 0, 1);
        gemm_wmma<5><<<g, NTHREADS, 0, stream>>>(kh, D_HEADC, projT, D_HEADC, nullptr, kpT,
                                                 0, D_HEADC, 0, 0, 0, 0, 0, 0, 1);
    }
    // 4) kv = k'^T @ v per (b,h): C[256,64], K=4096, stored transposed [B,H,64,256].
    {
        dim3 g(TWO_M / BM, D_HEADC / BN, BATCHC * N_HEADC);
        gemm_wmma<6><<<g, NTHREADS, 0, stream>>>(
            kpT, SEQC, vT, SEQC, nullptr, kvT, TWO_M, SEQC,
            /*sAB*/ (long)N_HEADC * TWO_M * SEQC, /*sAH*/ (long)TWO_M * SEQC,
            /*sBB*/ (long)N_HEADC * D_HEADC * SEQC, /*sBH*/ (long)D_HEADC * SEQC,
            /*sCB*/ (long)N_HEADC * D_HEADC * TWO_M, /*sCH*/ (long)D_HEADC * TWO_M,
            N_HEADC);
    }
    // 5) out = q' @ kv per (b,h): C[4096,64], K=256.
    {
        dim3 g(SEQC / BM, D_HEADC / BN, BATCHC * N_HEADC);
        gemm_wmma<2><<<g, NTHREADS, 0, stream>>>(
            qp, (long)N_HEADC * TWO_M, kvT, TWO_M, nullptr, attn, D_MODELC, TWO_M,
            /*sAB*/ (long)SEQC * N_HEADC * TWO_M, /*sAH*/ TWO_M,
            /*sBB*/ (long)N_HEADC * D_HEADC * TWO_M, /*sBH*/ (long)D_HEADC * TWO_M,
            /*sCB*/ (long)SEQC * D_MODELC, /*sCH*/ D_HEADC, N_HEADC);
    }
    // 6) final projection: d_out f32 = attn @ Wo^T + bo.
    {
        dim3 g(BLC / BM, D_MODELC / BN, 1);
        gemm_wmma<1><<<g, NTHREADS, 0, stream>>>(attn, D_MODELC, Woh, D_MODELC, bo,
                                                 (float*)d_out, D_MODELC, D_MODELC,
                                                 0, 0, 0, 0, 0, 0, 1);
    }
}
